// Transformer_63548336111828
// MI455X (gfx1250) — matrix operand order repaired and checked by reading, not compiled, not run on hardware
//
#include <hip/hip_runtime.h>
#include <math.h>

#define D_MODEL 512
#define D_FFN   2048
#define SEQ     2048
#define BATCH   4
#define NBLK    6
#define EPS     1e-5f

typedef __attribute__((ext_vector_type(16))) _Float16 v16h;
typedef __attribute__((ext_vector_type(8)))  _Float16 v8h;
typedef __attribute__((ext_vector_type(4)))  _Float16 v4h;
typedef __attribute__((ext_vector_type(8)))  float    v8f;
typedef __attribute__((ext_vector_type(4)))  float    v4f;

#define TM 128
#define TN 64
#define TK 32
#define LDA (TK + 8)   // 40 halfs = 80B row stride (rows stay 16B-aligned)

// C[M,N] = relu?( alpha * A[M,K] @ (TRANSB ? B[N,K]^T : B[K,N]) + bias[N] )
// batched via blockIdx.z with element strides sA/sB/sC.
// K%32==0, N%64==0 assumed. CHECK=1 adds M-edge guards (decoder M=4/M=1).
// Software-pipelined: global loads for tile kt+1 overlap WMMA on tile kt.
template<int TRANSB, int CHECK>
__global__ __launch_bounds__(256)
void gemm_wmma_f16(const float* __restrict__ A, const float* __restrict__ B,
                   const float* __restrict__ bias, float* __restrict__ C,
                   int M, int N, int K,
                   long sA, long sB, long sC,
                   float alpha, int relu)
{
    __shared__ __align__(16) _Float16 As[2][TM][LDA];  // [m][k]
    __shared__ __align__(16) _Float16 Bs[2][TN][LDA];  // [n][k]

    A += (long)blockIdx.z * sA;
    B += (long)blockIdx.z * sB;
    C += (long)blockIdx.z * sC;

    const int m0   = blockIdx.y * TM;
    const int n0   = blockIdx.x * TN;
    const int tid  = threadIdx.x;
    const int lane = tid & 31;
    const int wave = tid >> 5;          // 8 waves: 4 over M, 2 over N
    const int wm   = (wave & 3) * 32;
    const int wn   = (wave >> 2) * 32;
    const int mlo  = lane & 15;
    const int half = lane >> 4;
    const int nk   = K / TK;

    // ---- loop-invariant staging pointers: advance by TK columns per step ----
    const float* aPtr[4];
    _Float16*    aDst[4];
    bool         aZero[4];
    #pragma unroll
    for (int it = 0; it < 4; ++it) {
        int g = it * 256 + tid;                 // 0..1023
        int m = g >> 3, kk = (g & 7) * 4;
        int row = m0 + m;
        if (CHECK) { aZero[it] = row >= M; if (row > M - 1) row = M - 1; }
        else       { aZero[it] = false; }
        aPtr[it] = A + (long)row * K + kk;      // clamped: always safe to load
        aDst[it] = &As[0][m][kk];
    }
    const float* bPtr[2];
    _Float16*    bDst[2];
    #pragma unroll
    for (int it = 0; it < 2; ++it) {
        int g = it * 256 + tid;                 // 0..511
        if (TRANSB) {
            int n = g >> 3, kk = (g & 7) * 4;
            bPtr[it] = B + (long)(n0 + n) * K + kk;
            bDst[it] = &Bs[0][n][kk];
        } else {
            int kk = g >> 4, nn = (g & 15) * 4;
            bPtr[it] = B + (long)kk * N + (n0 + nn);
            bDst[it] = &Bs[0][nn][kk];
        }
    }
    const long bStep = TRANSB ? (long)TK : (long)TK * N;

    v8f acc[2][2] = {};
    v4f aReg[4], bReg[2];

    // issue global loads for one K-tile into registers (branch-free)
    auto globalLoad = [&]() {
        #pragma unroll
        for (int it = 0; it < 4; ++it) {
            v4f vv = *(const v4f*)aPtr[it];
            if (CHECK && aZero[it]) vv = (v4f){ 0.f, 0.f, 0.f, 0.f };
            aReg[it] = vv;
            aPtr[it] += TK;
        }
        #pragma unroll
        for (int it = 0; it < 2; ++it) {
            bReg[it] = *(const v4f*)bPtr[it];
            bPtr[it] += bStep;
        }
    };

    // convert f32 regs -> f16 and store into LDS buffer p
    auto ldsStore = [&](int p) {
        const int aoff = p * (TM * LDA);
        const int boff = p * (TN * LDA);
        #pragma unroll
        for (int it = 0; it < 4; ++it) {
            v4h hh = { (_Float16)aReg[it].x, (_Float16)aReg[it].y,
                       (_Float16)aReg[it].z, (_Float16)aReg[it].w };
            *(v4h*)(aDst[it] + aoff) = hh;
        }
        #pragma unroll
        for (int it = 0; it < 2; ++it) {
            if (TRANSB) {
                v4h hh = { (_Float16)bReg[it].x, (_Float16)bReg[it].y,
                           (_Float16)bReg[it].z, (_Float16)bReg[it].w };
                *(v4h*)(bDst[it] + boff) = hh;
            } else {
                _Float16* d = bDst[it] + boff;   // rows nn..nn+3, stride LDA
                d[0 * LDA] = (_Float16)bReg[it].x;
                d[1 * LDA] = (_Float16)bReg[it].y;
                d[2 * LDA] = (_Float16)bReg[it].z;
                d[3 * LDA] = (_Float16)bReg[it].w;
            }
        }
    };

    // WMMA on LDS buffer p (fragments per ISA 7.12.2, wave32)
    auto compute = [&](int p) {
        v16h af[2], bf[2];
        #pragma unroll
        for (int t = 0; t < 2; ++t) {
            const _Float16* ap = &As[p][wm + t * 16 + mlo][0];
            v8h alo = *(const v8h*)(ap + 8 * half);        // K = 8*half + 0..7
            v8h ahi = *(const v8h*)(ap + 16 + 8 * half);   // K = 16 + 8*half + 0..7
            af[t] = __builtin_shufflevector(alo, ahi,
                        0,1,2,3,4,5,6,7,8,9,10,11,12,13,14,15);
            const _Float16* bp = &Bs[p][wn + t * 16 + mlo][0];
            v8h blo = *(const v8h*)(bp + 8 * half);        // K = 8*half + 0..7
            v8h bhi = *(const v8h*)(bp + 16 + 8 * half);   // K = 16 + 8*half + 0..7
            bf[t] = __builtin_shufflevector(blo, bhi,
                        0,1,2,3,4,5,6,7,8,9,10,11,12,13,14,15);
        }
        #pragma unroll
        for (int i = 0; i < 2; ++i)
          #pragma unroll
          for (int j = 0; j < 2; ++j)
            acc[i][j] = __builtin_amdgcn_wmma_f32_16x16x32_f16(
                false, af[i], false, bf[j], (short)0, acc[i][j], false, false);
    };

    globalLoad();                         // prologue: tile 0 -> regs
    for (int kt = 0; kt < nk; ++kt) {
        const int p = kt & 1;
        ldsStore(p);                      // regs (tile kt) -> LDS buf p
        __syncthreads();                  // tile visible; also fences buf reuse
        if (kt + 1 < nk) {
            globalLoad();                 // tile kt+1 in flight during WMMA
            if (kt + 2 < nk)
                __builtin_prefetch(aPtr[0], 0, 1);   // global_prefetch_b8, tile kt+2
        }
        compute(p);                       // WMMA overlaps outstanding loads
    }

    // ---- epilogue: D layout n = lane%16, m = r + 8*half ----
    #pragma unroll
    for (int i = 0; i < 2; ++i) {
      #pragma unroll
      for (int j = 0; j < 2; ++j) {
        int nn = n0 + wn + j * 16 + mlo;
        float bv = bias ? bias[nn] : 0.f;
        #pragma unroll
        for (int r = 0; r < 8; ++r) {
            int mm = m0 + wm + i * 16 + r + 8 * half;
            if (!CHECK || mm < M) {
                float val = acc[i][j][r] * alpha + bv;
                if (relu) val = fmaxf(val, 0.f);
                C[(long)mm * N + nn] = val;
            }
        }
      }
    }
}

__global__ __launch_bounds__(256)
void softmax_rows(float* __restrict__ S, int cols)
{
    float* p = S + (long)blockIdx.x * cols;
    const int tid = threadIdx.x;
    const int lane = tid & 31, wave = tid >> 5;
    __shared__ float red[8];

    float m = -3.0e38f;
    for (int c = tid; c < cols; c += 256) m = fmaxf(m, p[c]);
    #pragma unroll
    for (int o = 16; o >= 1; o >>= 1) m = fmaxf(m, __shfl_xor(m, o, 32));
    if (lane == 0) red[wave] = m;
    __syncthreads();
    if (tid == 0) { float t = red[0]; for (int w = 1; w < 8; ++w) t = fmaxf(t, red[w]); red[0] = t; }
    __syncthreads();
    m = red[0];
    __syncthreads();

    float s = 0.f;
    for (int c = tid; c < cols; c += 256) { float e = __expf(p[c] - m); p[c] = e; s += e; }
    #pragma unroll
    for (int o = 16; o >= 1; o >>= 1) s += __shfl_xor(s, o, 32);
    if (lane == 0) red[wave] = s;
    __syncthreads();
    if (tid == 0) { float t = 0.f; for (int w = 0; w < 8; ++w) t += red[w]; red[0] = t; }
    __syncthreads();
    float inv = 1.f / red[0];
    for (int c = tid; c < cols; c += 256) p[c] *= inv;
}

// O[row] = g * normalize(X[row] + R[row]) + b  (row length D_MODEL=512, 2 elems/thread)
__global__ __launch_bounds__(256)
void add_layernorm(const float* __restrict__ X, const float* __restrict__ R,
                   const float* __restrict__ g, const float* __restrict__ b,
                   float* __restrict__ O)
{
    const long base = (long)blockIdx.x * D_MODEL;
    const int tid = threadIdx.x;
    const int lane = tid & 31, wave = tid >> 5;
    __shared__ float red[8];

    float x0 = X[base + tid]       + R[base + tid];
    float x1 = X[base + 256 + tid] + R[base + 256 + tid];

    float s = x0 + x1;
    #pragma unroll
    for (int o = 16; o >= 1; o >>= 1) s += __shfl_xor(s, o, 32);
    if (lane == 0) red[wave] = s;
    __syncthreads();
    if (tid == 0) { float t = 0.f; for (int w = 0; w < 8; ++w) t += red[w]; red[0] = t; }
    __syncthreads();
    float mu = red[0] * (1.0f / D_MODEL);
    __syncthreads();

    float d0 = x0 - mu, d1 = x1 - mu;
    float q = d0 * d0 + d1 * d1;
    #pragma unroll
    for (int o = 16; o >= 1; o >>= 1) q += __shfl_xor(q, o, 32);
    if (lane == 0) red[wave] = q;
    __syncthreads();
    if (tid == 0) { float t = 0.f; for (int w = 0; w < 8; ++w) t += red[w]; red[0] = t; }
    __syncthreads();
    float inv = rsqrtf(red[0] * (1.0f / D_MODEL) + EPS);

    O[base + tid]       = g[tid]       * d0 * inv + b[tid];
    O[base + 256 + tid] = g[tid + 256] * d1 * inv + b[tid + 256];
}

__global__ __launch_bounds__(256)
void enc_embed(const int* __restrict__ x, const float* __restrict__ emb,
               float* __restrict__ out)
{
    long idx = (long)blockIdx.x * 256 + threadIdx.x;      // flat [B*S*D]
    if (idx >= (long)BATCH * SEQ * D_MODEL) return;
    int d    = (int)(idx & (D_MODEL - 1));
    long bs  = idx >> 9;                                  // D_MODEL == 512
    int spos = (int)(bs & (SEQ - 1));
    int tok  = x[bs];
    const float kln = -9.2103403719761836f / (float)D_MODEL;   // -ln(10000)/D
    float div = __expf((float)(d & ~1) * kln);
    float ang = (float)spos * div;
    float pe  = (d & 1) ? __cosf(ang) : __sinf(ang);
    out[idx] = emb[(long)tok * D_MODEL + d] * 22.62741699796952f + pe; // sqrt(512)
}

__global__ __launch_bounds__(256)
void dec_embed(const int* __restrict__ tgt, const float* __restrict__ emb,
               float* __restrict__ out)
{
    int idx = blockIdx.x * 256 + threadIdx.x;             // [B*D]
    if (idx >= BATCH * D_MODEL) return;
    int d = idx & (D_MODEL - 1);
    int b = idx >> 9;
    float pe = (d & 1) ? 1.0f : 0.0f;                     // pos-encoding at s=0
    out[idx] = emb[(long)tgt[b] * D_MODEL + d] * 22.62741699796952f + pe;
}

__global__ __launch_bounds__(256)
void copy_f32(const float* __restrict__ src, float* __restrict__ dst, int n)
{
    int i = blockIdx.x * 256 + threadIdx.x;
    if (i < n) dst[i] = src[i];
}

// host-side dispatcher: picks the branch-free variant whenever M tiles evenly
static void run_gemm(hipStream_t stream, dim3 grid,
                     const float* A, const float* B, const float* bias, float* C,
                     int M, int N, int K, long sA, long sB, long sC,
                     float alpha, int transB, int relu)
{
    const dim3 blk(256);
    const bool check = (M % TM) != 0;
    if (transB) {
        if (check) gemm_wmma_f16<1,1><<<grid, blk, 0, stream>>>(A,B,bias,C,M,N,K,sA,sB,sC,alpha,relu);
        else       gemm_wmma_f16<1,0><<<grid, blk, 0, stream>>>(A,B,bias,C,M,N,K,sA,sB,sC,alpha,relu);
    } else {
        if (check) gemm_wmma_f16<0,1><<<grid, blk, 0, stream>>>(A,B,bias,C,M,N,K,sA,sB,sC,alpha,relu);
        else       gemm_wmma_f16<0,0><<<grid, blk, 0, stream>>>(A,B,bias,C,M,N,K,sA,sB,sC,alpha,relu);
    }
}

extern "C" void kernel_launch(void* const* d_in, const int* in_sizes, int n_in,
                              void* d_out, int out_size, void* d_ws, size_t ws_size,
                              hipStream_t stream)
{
    (void)in_sizes; (void)n_in; (void)out_size; (void)ws_size;

    const int*   x       = (const int*)d_in[0];
    const int*   target  = (const int*)d_in[1];
    const float* in_emb  = (const float*)d_in[2];
    const float* out_emb = (const float*)d_in[3];

    const float* eWq  = (const float*)d_in[4];   const float* ebq  = (const float*)d_in[5];
    const float* eWk  = (const float*)d_in[6];   const float* ebk  = (const float*)d_in[7];
    const float* eWv  = (const float*)d_in[8];   const float* ebv  = (const float*)d_in[9];
    const float* eln1g= (const float*)d_in[10];  const float* eln1b= (const float*)d_in[11];
    const float* eW1  = (const float*)d_in[12];  const float* eb1  = (const float*)d_in[13];
    const float* eW2  = (const float*)d_in[14];  const float* eb2  = (const float*)d_in[15];
    const float* eln2g= (const float*)d_in[16];  const float* eln2b= (const float*)d_in[17];

    const float* dWq1 = (const float*)d_in[18];  const float* dbq1 = (const float*)d_in[19];
    const float* dWk1 = (const float*)d_in[20];  const float* dbk1 = (const float*)d_in[21];
    const float* dWv1 = (const float*)d_in[22];  const float* dbv1 = (const float*)d_in[23];
    const float* dln1g= (const float*)d_in[24];  const float* dln1b= (const float*)d_in[25];
    const float* dWq2 = (const float*)d_in[26];  const float* dbq2 = (const float*)d_in[27];
    const float* dWk2 = (const float*)d_in[28];  const float* dbk2 = (const float*)d_in[29];
    const float* dWv2 = (const float*)d_in[30];  const float* dbv2 = (const float*)d_in[31];
    const float* dln2g= (const float*)d_in[32];  const float* dln2b= (const float*)d_in[33];
    const float* dW1  = (const float*)d_in[34];  const float* db1  = (const float*)d_in[35];
    const float* dW2  = (const float*)d_in[36];  const float* db2  = (const float*)d_in[37];
    const float* dln3g= (const float*)d_in[38];  const float* dln3b= (const float*)d_in[39];
    (void)dWq1; (void)dbq1; (void)dWk1; (void)dbk1;  // unused: 1-token causal attn == V proj

    float* ws = (float*)d_ws;
    const long TOKS = (long)BATCH * SEQ;      // 8192
    const long ACT  = TOKS * D_MODEL;         // 4,194,304
    float* h    = ws;                         // encoder hidden / enc_out
    float* q    = h + ACT;
    float* k    = q + ACT;
    float* v    = k + ACT;
    float* t    = v + ACT;
    float* big  = t + ACT;                    // scores [4,2048,2048] / ffn mid [8192,2048]
    float* dh   = big + TOKS * D_FFN;
    float* dq   = dh  + BATCH * D_MODEL;
    float* dt   = dq  + BATCH * D_MODEL;
    float* dsc  = dt  + BATCH * D_MODEL;
    float* dmid = dsc + (long)BATCH * SEQ;

    const float ALPHA_SM = 0.044194173824159216f;   // 1/sqrt(512)
    const dim3 blk(256);

    enc_embed<<<(unsigned)((ACT + 255) / 256), blk, 0, stream>>>(x, in_emb, h);

    for (int i = 0; i < NBLK; ++i) {
        const long W   = (long)i * D_MODEL * D_MODEL;
        const long Bv  = (long)i * D_MODEL;
        const long Wf  = (long)i * D_MODEL * D_FFN;
        const long Bf  = (long)i * D_FFN;

        dim3 gp(D_MODEL / TN, (unsigned)(TOKS / TM), 1);
        run_gemm(stream, gp, h, eWq + W, ebq + Bv, q,
                 (int)TOKS, D_MODEL, D_MODEL, 0, 0, 0, 1.f, 0, 0);
        run_gemm(stream, gp, h, eWk + W, ebk + Bv, k,
                 (int)TOKS, D_MODEL, D_MODEL, 0, 0, 0, 1.f, 0, 0);
        run_gemm(stream, gp, h, eWv + W, ebv + Bv, v,
                 (int)TOKS, D_MODEL, D_MODEL, 0, 0, 0, 1.f, 0, 0);

        dim3 gs(SEQ / TN, SEQ / TM, BATCH);             // scores = Q @ K^T / sqrt(d)
        run_gemm(stream, gs, q, k, nullptr, big,
                 SEQ, SEQ, D_MODEL, (long)SEQ * D_MODEL, (long)SEQ * D_MODEL,
                 (long)SEQ * SEQ, ALPHA_SM, 1, 0);

        softmax_rows<<<(unsigned)TOKS, blk, 0, stream>>>(big, SEQ);

        dim3 ga(D_MODEL / TN, SEQ / TM, BATCH);         // ctx = P @ V
        run_gemm(stream, ga, big, v, nullptr, t,
                 SEQ, D_MODEL, SEQ, (long)SEQ * SEQ, (long)SEQ * D_MODEL,
                 (long)SEQ * D_MODEL, 1.f, 0, 0);

        add_layernorm<<<(unsigned)TOKS, blk, 0, stream>>>(t, h, eln1g + Bv, eln1b + Bv, h);

        dim3 gf1(D_FFN / TN, (unsigned)(TOKS / TM), 1);
        run_gemm(stream, gf1, h, eW1 + Wf, eb1 + Bf, big,
                 (int)TOKS, D_FFN, D_MODEL, 0, 0, 0, 1.f, 0, 1);
        dim3 gf2(D_MODEL / TN, (unsigned)(TOKS / TM), 1);
        run_gemm(stream, gf2, big, eW2 + Wf, eb2 + Bv, t,
                 (int)TOKS, D_MODEL, D_FFN, 0, 0, 0, 1.f, 0, 0);

        add_layernorm<<<(unsigned)TOKS, blk, 0, stream>>>(t, h, eln2g + Bv, eln2b + Bv, h);
    }

    // ---------------- decoder (1 token per batch) ----------------
    dec_embed<<<(BATCH * D_MODEL + 255) / 256, blk, 0, stream>>>(target, out_emb, dh);

    for (int i = 0; i < NBLK; ++i) {
        const long W  = (long)i * D_MODEL * D_MODEL;
        const long Bv = (long)i * D_MODEL;
        const long Wf = (long)i * D_MODEL * D_FFN;
        const long Bf = (long)i * D_FFN;

        dim3 g4(D_MODEL / TN, 1, 1);
        // 1-token causal self-attention == V projection (softmax of 1x1 is 1)
        run_gemm(stream, g4, dh, dWv1 + W, dbv1 + Bv, dt,
                 BATCH, D_MODEL, D_MODEL, 0, 0, 0, 1.f, 0, 0);
        add_layernorm<<<BATCH, blk, 0, stream>>>(dt, dh, dln1g + Bv, dln1b + Bv, dh);

        // cross-attention over enc_out
        dim3 gp(D_MODEL / TN, (unsigned)(TOKS / TM), 1);
        run_gemm(stream, gp, h, dWk2 + W, dbk2 + Bv, k,
                 (int)TOKS, D_MODEL, D_MODEL, 0, 0, 0, 1.f, 0, 0);
        run_gemm(stream, gp, h, dWv2 + W, dbv2 + Bv, v,
                 (int)TOKS, D_MODEL, D_MODEL, 0, 0, 0, 1.f, 0, 0);
        run_gemm(stream, g4, dh, dWq2 + W, dbq2 + Bv, dq,
                 BATCH, D_MODEL, D_MODEL, 0, 0, 0, 1.f, 0, 0);

        dim3 gsc(SEQ / TN, 1, BATCH);
        run_gemm(stream, gsc, dq, k, nullptr, dsc,
                 1, SEQ, D_MODEL, (long)D_MODEL, (long)SEQ * D_MODEL, (long)SEQ,
                 ALPHA_SM, 1, 0);
        softmax_rows<<<BATCH, blk, 0, stream>>>(dsc, SEQ);
        dim3 gao(D_MODEL / TN, 1, BATCH);
        run_gemm(stream, gao, dsc, v, nullptr, dt,
                 1, D_MODEL, SEQ, (long)SEQ, (long)SEQ * D_MODEL, (long)D_MODEL,
                 1.f, 0, 0);
        add_layernorm<<<BATCH, blk, 0, stream>>>(dt, dh, dln2g + Bv, dln2b + Bv, dh);

        dim3 gm(D_FFN / TN, 1, 1);
        run_gemm(stream, gm, dh, dW1 + Wf, db1 + Bf, dmid,
                 BATCH, D_FFN, D_MODEL, 0, 0, 0, 1.f, 0, 1);
        run_gemm(stream, g4, dmid, dW2 + Wf, db2 + Bv, dt,
                 BATCH, D_MODEL, D_FFN, 0, 0, 0, 1.f, 0, 0);
        add_layernorm<<<BATCH, blk, 0, stream>>>(dt, dh, dln3g + Bv, dln3b + Bv, dh);
    }

    copy_f32<<<(BATCH * D_MODEL + 255) / 256, blk, 0, stream>>>(
        dh, (float*)d_out, BATCH * D_MODEL);
}
